// AttentionBlock_48730698941054
// MI455X (gfx1250) — matrix operand order repaired and checked by reading, not compiled, not run on hardware
//
#include <hip/hip_runtime.h>
#include <hip/hip_bf16.h>

typedef __attribute__((ext_vector_type(16))) __bf16 v16bf;
typedef __attribute__((ext_vector_type(8)))  __bf16 v8bf;
typedef __attribute__((ext_vector_type(8)))  float  v8f;

#define C_DIM 256
#define N_PIX 4096
#define HEADS 8
#define HD    32

// ---- WMMA helpers -------------------------------------------------------
__device__ __forceinline__ v8f wmma_bf16(v16bf a, v16bf b, v8f c) {
    return __builtin_amdgcn_wmma_f32_16x16x32_bf16(
        /*neg_a=*/false, a, /*neg_b=*/false, b,
        /*c_mod=*/(short)0, c, /*reuse_a=*/false, /*reuse_b=*/false);
}

// A fragment (16x32 bf16), source row-major (row, K), ld in elements.
// lane L<16: row=L, K = k0+0..7 and k0+16..23 ; lanes 16-31: K = k0+8..15, k0+24..31
__device__ __forceinline__ v16bf load_a16(const __bf16* __restrict__ base, int ld,
                                          int row0, int k0, int lane) {
    const int half = lane >> 4, r = lane & 15;
    const __bf16* p = base + (long)(row0 + r) * ld + k0 + half * 8;
    v8bf lo = *(const v8bf*)p;
    v8bf hi = *(const v8bf*)(p + 16);
    v16bf out = {};
#pragma unroll
    for (int i = 0; i < 8; i++) { out[i] = lo[i]; out[i + 8] = hi[i]; }
    return out;
}

// B fragment (32x16 bf16). Source stored as (col, K) row-major:
// lane L<16: col=L, K=k0+0..7 and k0+16..23 ; lanes 16-31: K=k0+8..15, k0+24..31
__device__ __forceinline__ v16bf load_b16(const __bf16* __restrict__ base, int ld,
                                          int col0, int k0, int lane) {
    const int half = lane >> 4, r = lane & 15;
    const __bf16* p = base + (long)(col0 + r) * ld + k0 + half * 8;
    v8bf lo = *(const v8bf*)p;
    v8bf hi = *(const v8bf*)(p + 16);
    v16bf out = {};
#pragma unroll
    for (int i = 0; i < 8; i++) { out[i] = lo[i]; out[i + 8] = hi[i]; }
    return out;
}

// ---- 0) convert weights to bf16 ----------------------------------------
__global__ void cvt_kernel(const float* __restrict__ qkvw, const float* __restrict__ projw,
                           __bf16* __restrict__ wq, __bf16* __restrict__ wp) {
    int i = blockIdx.x * blockDim.x + threadIdx.x;
    if (i < 768 * 256) wq[i] = (__bf16)qkvw[i];
    int j = i - 768 * 256;
    if (j >= 0 && j < 256 * 256) wp[j] = (__bf16)projw[j];
}

// ---- 1) GroupNorm, writes hT (b, n, c) bf16 ----------------------------
__global__ void gn_kernel(const float* __restrict__ x, const float* __restrict__ gamma,
                          const float* __restrict__ beta, __bf16* __restrict__ hT) {
    const int b = blockIdx.x >> 5;   // 64 blocks = 2 batches * 32 groups
    const int g = blockIdx.x & 31;
    const int tid = threadIdx.x;
    __shared__ float ssum[256], ssq[256];
    const float* xb = x + ((long)(b * C_DIM + g * 8)) * N_PIX;
    float s = 0.f, q = 0.f;
    for (int i = tid; i < 8 * N_PIX; i += 256) { float v = xb[i]; s += v; q += v * v; }
    ssum[tid] = s; ssq[tid] = q;
    __syncthreads();
    for (int st = 128; st > 0; st >>= 1) {
        if (tid < st) { ssum[tid] += ssum[tid + st]; ssq[tid] += ssq[tid + st]; }
        __syncthreads();
    }
    const float mean = ssum[0] * (1.0f / 32768.0f);
    const float var  = ssq[0] * (1.0f / 32768.0f) - mean * mean;
    const float rinv = rsqrtf(var + 1e-5f);
    for (int i = tid; i < 8 * N_PIX; i += 256) {
        int ci = i >> 12, n = i & 4095, c = g * 8 + ci;
        float v = (xb[i] - mean) * rinv * gamma[c] + beta[c];
        hT[((long)(b * N_PIX + n)) * C_DIM + c] = (__bf16)v;
    }
}

// ---- 2) QKV GEMM, 32x32 tile per wave (2x2 wmma tiling) -----------------
__global__ void qkv_kernel(const __bf16* __restrict__ wqkv, const float* __restrict__ qkvb,
                           const __bf16* __restrict__ hT,
                           __bf16* __restrict__ qO, __bf16* __restrict__ kO,
                           __bf16* __restrict__ vO) {
    const int lane = threadIdx.x & 31;
    const int w = blockIdx.x * (blockDim.x >> 5) + (threadIdx.x >> 5);
    const int b = w / (24 * 128);                 // 768/32=24 row blocks, 4096/32=128 col blocks
    const int rem = w % (24 * 128);
    const int o0 = (rem / 128) * 32, n0 = (rem & 127) * 32;
    const __bf16* hb = hT + (long)b * N_PIX * C_DIM;
    v8f acc00 = {}, acc01 = {}, acc10 = {}, acc11 = {};
#pragma unroll
    for (int k = 0; k < C_DIM; k += 32) {
        v16bf a0 = load_a16(wqkv, C_DIM, o0,      k, lane);
        v16bf a1 = load_a16(wqkv, C_DIM, o0 + 16, k, lane);
        v16bf b0 = load_b16(hb,   C_DIM, n0,      k, lane);
        v16bf b1 = load_b16(hb,   C_DIM, n0 + 16, k, lane);
        acc00 = wmma_bf16(a0, b0, acc00);
        acc01 = wmma_bf16(a0, b1, acc01);
        acc10 = wmma_bf16(a1, b0, acc10);
        acc11 = wmma_bf16(a1, b1, acc11);
    }
    const int half = lane >> 4, col = lane & 15;
#pragma unroll
    for (int i = 0; i < 2; i++) {
#pragma unroll
        for (int j = 0; j < 2; j++) {
            const v8f acc = (i == 0) ? (j == 0 ? acc00 : acc01) : (j == 0 ? acc10 : acc11);
            const int n = n0 + j * 16 + col;
#pragma unroll
            for (int r = 0; r < 8; r++) {
                int o = o0 + i * 16 + r + half * 8;
                float val = acc[r] + qkvb[o];
                int c = o & 255, head = c >> 5, d = c & 31;
                if (o < 256)
                    qO[(((long)(b * HEADS + head)) * N_PIX + n) * HD + d] = (__bf16)val;
                else if (o < 512)
                    kO[(((long)(b * HEADS + head)) * N_PIX + n) * HD + d] = (__bf16)val;
                else
                    vO[(((long)(b * HEADS + head)) * HD + d) * N_PIX + n] = (__bf16)val;
            }
        }
    }
}

// ---- 3) flash attention: one 32-row q block per wave --------------------
__global__ void attn_kernel(const __bf16* __restrict__ qO, const __bf16* __restrict__ kO,
                            const __bf16* __restrict__ vO, __bf16* __restrict__ attnT) {
    __shared__ __align__(32) __bf16 Pb[4][32 * 32];
    const int lane = threadIdx.x & 31;
    const int wib = threadIdx.x >> 5;
    const int w = blockIdx.x * 4 + wib;
    const int b = w / (HEADS * 128);              // 4096/32 = 128 q blocks per head
    const int rem = w % (HEADS * 128);
    const int head = rem >> 7;
    const int n0 = (rem & 127) * 32;
    const __bf16* qb = qO + ((long)(b * HEADS + head)) * N_PIX * HD;
    const __bf16* kb = kO + ((long)(b * HEADS + head)) * N_PIX * HD;
    const __bf16* vb = vO + ((long)(b * HEADS + head)) * HD * N_PIX;
    const float scale = 0.17677669529663687f;     // 1/sqrt(32)

    v16bf qa0 = load_a16(qb, HD, n0,      0, lane);
    v16bf qa1 = load_a16(qb, HD, n0 + 16, 0, lane);
    float mrow[2][8], lrow[2][8];
    v8f acc[2][2] = {{{}, {}}, {{}, {}}};
#pragma unroll
    for (int qi = 0; qi < 2; qi++)
#pragma unroll
        for (int r = 0; r < 8; r++) { mrow[qi][r] = -3.0e38f; lrow[qi][r] = 0.f; }
    const int half = lane >> 4, col = lane & 15;
    __bf16* P = &Pb[wib][0];
    const v8f zf = {};

    for (int m0 = 0; m0 < N_PIX; m0 += 32) {
        v16bf bk0 = load_b16(kb, HD, m0,      0, lane);
        v16bf bk1 = load_b16(kb, HD, m0 + 16, 0, lane);
        v8f s[2][2];
        s[0][0] = wmma_bf16(qa0, bk0, zf);
        s[0][1] = wmma_bf16(qa0, bk1, zf);
        s[1][0] = wmma_bf16(qa1, bk0, zf);
        s[1][1] = wmma_bf16(qa1, bk1, zf);
#pragma unroll
        for (int qi = 0; qi < 2; qi++) {
#pragma unroll
            for (int r = 0; r < 8; r++) {
                float a  = s[qi][0][r] * scale;
                float c2 = s[qi][1][r] * scale;
                float t = fmaxf(a, c2);
#pragma unroll
                for (int msk = 1; msk < 16; msk <<= 1) t = fmaxf(t, __shfl_xor(t, msk, 32));
                float mn = fmaxf(mrow[qi][r], t);
                float corr = __expf(mrow[qi][r] - mn);
                float p0 = __expf(a - mn), p1 = __expf(c2 - mn);
                float su = p0 + p1;
#pragma unroll
                for (int msk = 1; msk < 16; msk <<= 1) su += __shfl_xor(su, msk, 32);
                lrow[qi][r] = lrow[qi][r] * corr + su;
                acc[qi][0][r] *= corr; acc[qi][1][r] *= corr;
                mrow[qi][r] = mn;
                int row = qi * 16 + r + half * 8;
                P[row * 32 + col]      = (__bf16)p0;
                P[row * 32 + 16 + col] = (__bf16)p1;
            }
        }
        asm volatile("s_wait_dscnt 0" ::: "memory");
        v16bf pa[2];
#pragma unroll
        for (int qi = 0; qi < 2; qi++) {
            pa[qi] = (v16bf){};
            const __bf16* pp = P + (long)(qi * 16 + (lane & 15)) * 32 + (lane >> 4) * 8;
#pragma unroll
            for (int i = 0; i < 8; i++) { pa[qi][i] = pp[i]; pa[qi][i + 8] = pp[16 + i]; }
        }
        asm volatile("" ::: "memory");
        v16bf bv0 = load_b16(vb, N_PIX, 0,  m0, lane);
        v16bf bv1 = load_b16(vb, N_PIX, 16, m0, lane);
        acc[0][0] = wmma_bf16(pa[0], bv0, acc[0][0]);
        acc[0][1] = wmma_bf16(pa[0], bv1, acc[0][1]);
        acc[1][0] = wmma_bf16(pa[1], bv0, acc[1][0]);
        acc[1][1] = wmma_bf16(pa[1], bv1, acc[1][1]);
    }
#pragma unroll
    for (int qi = 0; qi < 2; qi++) {
#pragma unroll
        for (int r = 0; r < 8; r++) {
            int n = n0 + qi * 16 + r + half * 8;
            float inv = 1.0f / lrow[qi][r];
            long basei = ((long)(b * N_PIX + n)) * C_DIM + head * HD;
            attnT[basei + col]      = (__bf16)(acc[qi][0][r] * inv);
            attnT[basei + 16 + col] = (__bf16)(acc[qi][1][r] * inv);
        }
    }
}

// ---- 4) proj GEMM + bias + residual, 32x32 tile per wave ----------------
__global__ void proj_kernel(const __bf16* __restrict__ wproj, const float* __restrict__ projb,
                            const __bf16* __restrict__ attnT, const float* __restrict__ x,
                            float* __restrict__ out) {
    const int lane = threadIdx.x & 31;
    const int w = blockIdx.x * (blockDim.x >> 5) + (threadIdx.x >> 5);
    const int b = w / (8 * 128);                  // 256/32=8 row blocks, 128 col blocks
    const int rem = w % (8 * 128);
    const int o0 = (rem / 128) * 32, n0 = (rem & 127) * 32;
    const __bf16* ab = attnT + (long)b * N_PIX * C_DIM;
    v8f acc00 = {}, acc01 = {}, acc10 = {}, acc11 = {};
#pragma unroll
    for (int k = 0; k < C_DIM; k += 32) {
        v16bf a0 = load_a16(wproj, C_DIM, o0,      k, lane);
        v16bf a1 = load_a16(wproj, C_DIM, o0 + 16, k, lane);
        v16bf b0 = load_b16(ab,    C_DIM, n0,      k, lane);
        v16bf b1 = load_b16(ab,    C_DIM, n0 + 16, k, lane);
        acc00 = wmma_bf16(a0, b0, acc00);
        acc01 = wmma_bf16(a0, b1, acc01);
        acc10 = wmma_bf16(a1, b0, acc10);
        acc11 = wmma_bf16(a1, b1, acc11);
    }
    const int half = lane >> 4, coln = lane & 15;
#pragma unroll
    for (int i = 0; i < 2; i++) {
#pragma unroll
        for (int j = 0; j < 2; j++) {
            const v8f acc = (i == 0) ? (j == 0 ? acc00 : acc01) : (j == 0 ? acc10 : acc11);
            const int n = n0 + j * 16 + coln;
#pragma unroll
            for (int r = 0; r < 8; r++) {
                int o = o0 + i * 16 + r + half * 8;
                long idx = ((long)(b * C_DIM + o)) * N_PIX + n;
                out[idx] = x[idx] + acc[r] + projb[o];
            }
        }
    }
}

extern "C" void kernel_launch(void* const* d_in, const int* in_sizes, int n_in,
                              void* d_out, int out_size, void* d_ws, size_t ws_size,
                              hipStream_t stream) {
    const float* x      = (const float*)d_in[0];
    const float* gamma  = (const float*)d_in[1];
    const float* beta   = (const float*)d_in[2];
    const float* qkvw   = (const float*)d_in[3];
    const float* qkvb   = (const float*)d_in[4];
    const float* projw  = (const float*)d_in[5];
    const float* projb  = (const float*)d_in[6];
    float* out = (float*)d_out;

    char* ws = (char*)d_ws;
    __bf16* wq    = (__bf16*)(ws + 0);                    // 768*256
    __bf16* wp    = (__bf16*)(ws + 393216);               // 256*256
    __bf16* hT    = (__bf16*)(ws + 524288);               // 2*4096*256
    __bf16* qO    = (__bf16*)(ws + 4718592);              // 2*8*4096*32
    __bf16* kO    = (__bf16*)(ws + 8912896);
    __bf16* vO    = (__bf16*)(ws + 13107200);
    __bf16* attnT = (__bf16*)(ws + 17301504);             // 2*4096*256

    cvt_kernel<<<1024, 256, 0, stream>>>(qkvw, projw, wq, wp);
    gn_kernel<<<64, 256, 0, stream>>>(x, gamma, beta, hT);
    // 2 * 24 * 128 = 6144 waves, 8 waves / 256-thread block
    qkv_kernel<<<768, 256, 0, stream>>>(wq, qkvb, hT, qO, kO, vO);
    // 2 * 8 * 128 = 2048 waves, 4 waves / 128-thread block
    attn_kernel<<<512, 128, 0, stream>>>(qO, kO, vO, attnT);
    // 2 * 8 * 128 = 2048 waves, 8 waves / 256-thread block
    proj_kernel<<<256, 256, 0, stream>>>(wp, projb, attnT, x, out);
}
